// KalmanNetArch1_70901320123078
// MI455X (gfx1250) — matrix operand order repaired and checked by reading, not compiled, not run on hardware
//
#include <hip/hip_runtime.h>
#include <hip/hip_bf16.h>
#include <math.h>
#include <stdint.h>

typedef __bf16 bf16;
typedef __attribute__((ext_vector_type(16))) __bf16 v16bf;
typedef __attribute__((ext_vector_type(8)))  __bf16 v8bf;
typedef __attribute__((ext_vector_type(8)))  float  v8f;

#define BATCH   4096
#define MD      10          // state dim m
#define ND      10          // obs dim n
#define HD      2000        // GRU hidden
#define DINF    40          // feature dim
#define KPAD    2016        // 63 * 32, padded K for WMMA loops
#define KPAD_IN 64          // padded DIN (2 * 32)
#define NKSTEP  63          // KPAD / 32
#define EPS_C   0.001f

// ---------------- workspace layout (bytes, 256-aligned) ----------------
#define OFF_FEATS 0ull                      // bf16 4096 x 64          = 524288
#define OFF_WIN   524288ull                 // bf16 2000 x 64          = 256000
#define OFF_X     780288ull                 // bf16 4096 x 2016        = 16515072
#define OFF_H0    17295360ull               // bf16 4096 x 2016
#define OFF_H1    33810432ull               // bf16 4096 x 2016
#define OFF_WIH   50325504ull               // bf16 6000 x 2016        = 24192000
#define OFF_WHH   74517504ull               // bf16 6000 x 2016
#define OFF_WOUT  98709504ull               // bf16 112  x 2016        = 451584
#define OFF_PRIOR 99161088ull               // f32  4096 x 10          = 163840
#define OFF_DY    99324928ull               // f32  4096 x 10
#define OFF_KG    99488768ull               // f32  4096 x 100         = 1638400

// ---------------- WMMA helpers ----------------
__device__ __forceinline__ v8f wmma_bf16(v16bf a, v16bf b, v8f c) {
  return __builtin_amdgcn_wmma_f32_16x16x32_bf16(false, a, false, b, (short)0, c,
                                                 false, false);
}

// A fragment (16x32, bf16), source row-major M x K.
__device__ __forceinline__ v16bf load_fragA(const bf16* __restrict__ base, int ld,
                                            int row0, int k0, int lane) {
  int r = lane & 15;
  int koff = (lane >> 4) * 16;
  const bf16* p = base + (size_t)(row0 + r) * ld + k0 + koff;
  v8bf lo = *(const v8bf*)p;
  v8bf hi = *(const v8bf*)(p + 8);
  return __builtin_shufflevector(lo, hi, 0, 1, 2, 3, 4, 5, 6, 7,
                                 8, 9, 10, 11, 12, 13, 14, 15);
}

// B fragment (32x16, bf16) from row-major N x K storage (global).
__device__ __forceinline__ v16bf load_fragB(const bf16* __restrict__ base, int ld,
                                            int row0, int k0, int lane) {
  int nn = lane & 15;
  int koff = (lane >> 4) * 16;
  const bf16* p = base + (size_t)(row0 + nn) * ld + k0 + koff;
  v8bf lo = *(const v8bf*)p;
  v8bf hi = *(const v8bf*)(p + 8);
  return __builtin_shufflevector(lo, hi, 0, 1, 2, 3, 4, 5, 6, 7,
                                 8, 9, 10, 11, 12, 13, 14, 15);
}

// B fragment from an LDS-resident 16x32 tile (row stride 32 elements).
__device__ __forceinline__ v16bf lds_fragB(const bf16* fr, int lane) {
  int nn = lane & 15;
  int koff = (lane >> 4) * 16;
  const v8bf* p = (const v8bf*)(fr + nn * 32 + koff);
  v8bf lo = p[0];
  v8bf hi = p[1];
  return __builtin_shufflevector(lo, hi, 0, 1, 2, 3, 4, 5, 6, 7,
                                 8, 9, 10, 11, 12, 13, 14, 15);
}

__device__ __forceinline__ float sigmoidf_(float x) {
  return 1.0f / (1.0f + __expf(-x));
}

#if __has_builtin(__builtin_amdgcn_tensor_load_to_lds)
#define HAS_TDM 1
typedef unsigned int v4u __attribute__((ext_vector_type(4)));
typedef int v8i __attribute__((ext_vector_type(8)));
typedef int v4i __attribute__((ext_vector_type(4)));

// TDM: load a 3D tile (x=32 K-elems, y=16 rows, z=3 gate blocks 2000 rows apart)
// from a (6000 x KPAD) bf16 weight matrix into LDS (3 KB, contiguous [z][y][x]).
__device__ __forceinline__ void tdm_load_w_tile(const bf16* gsrc, unsigned lds_off) {
  unsigned long long ga = (unsigned long long)(uintptr_t)gsrc;
  const unsigned d1s = 2000u * (unsigned)KPAD;   // tensor_dim1_stride = 4032000
  v4u g0;
  g0.x = 1u;                                     // count=1, user mode
  g0.y = lds_off;                                // lds_addr (bytes)
  g0.z = (unsigned)(ga & 0xFFFFFFFFu);           // global_addr[31:0]
  g0.w = (unsigned)((ga >> 32) & 0x01FFFFFFu) | (2u << 30);  // addr[56:32] | type=2
  v8i g1;
  g1[0] = 0x00010000;                                   // data_size=1 (2 bytes)
  g1[1] = (int)(((unsigned)KPAD & 0xFFFFu) << 16);      // tensor_dim0 lo16
  g1[2] = (int)((((unsigned)KPAD >> 16) & 0xFFFFu) |
                ((6000u & 0xFFFFu) << 16));             // dim0 hi | dim1 lo
  g1[3] = (int)((6000u >> 16) | (32u << 16));           // dim1 hi | tile_dim0=32
  g1[4] = (int)(16u | (3u << 16));                      // tile_dim1=16 | tile_dim2=3
  g1[5] = (int)(unsigned)KPAD;                          // tensor_dim0_stride lo32
  g1[6] = (int)((d1s & 0xFFFFu) << 16);                 // d0s hi16=0 | d1s lo16
  g1[7] = (int)(d1s >> 16);                             // d1s hi32
  v4i g2;
  g2[0] = 3;  g2[1] = 0;  g2[2] = 0;  g2[3] = 0;        // tensor_dim2=3
  v4i g3;
  g3[0] = 0;  g3[1] = 0;  g3[2] = 0;  g3[3] = 0;
  v8i g4;
  g4[0] = 0; g4[1] = 0; g4[2] = 0; g4[3] = 0;
  g4[4] = 0; g4[5] = 0; g4[6] = 0; g4[7] = 0;
  __builtin_amdgcn_tensor_load_to_lds(g0, g1, g2, g3, g4, 0);
}
#else
#define HAS_TDM 0
#endif

// ---------------- prep: priors, feats (bf16, padded to 64), dy ----------------
__global__ void prep_kernel(const float* __restrict__ y,
                            const float* __restrict__ post,
                            const float* __restrict__ post_prev,
                            const float* __restrict__ prior_prev,
                            const float* __restrict__ y_prev,
                            const float* __restrict__ F,
                            const float* __restrict__ Hm,
                            float* __restrict__ prior_out,
                            float* __restrict__ dy_out,
                            bf16* __restrict__ feats) {
  int b = blockIdx.x * blockDim.x + threadIdx.x;
  if (b >= BATCH) return;
  float yv[ND], pv[MD], prior[MD], ypred[ND];
#pragma unroll
  for (int i = 0; i < ND; ++i) yv[i] = y[b * ND + i];
#pragma unroll
  for (int i = 0; i < MD; ++i) pv[i] = post[b * MD + i];
#pragma unroll
  for (int i = 0; i < MD; ++i) {
    float s = 0.f;
#pragma unroll
    for (int j = 0; j < MD; ++j) s += F[i * MD + j] * pv[j];
    prior[i] = s;
  }
#pragma unroll
  for (int i = 0; i < ND; ++i) {
    float s = 0.f;
#pragma unroll
    for (int j = 0; j < MD; ++j) s += Hm[i * MD + j] * prior[j];
    ypred[i] = s;
  }
  float obs[ND], inn[ND], fev[MD], fup[MD];
  float n0 = 0.f, n1 = 0.f, n2 = 0.f, n3 = 0.f;
#pragma unroll
  for (int i = 0; i < ND; ++i) {
    obs[i] = yv[i] - y_prev[b * ND + i];   n0 += obs[i] * obs[i];
    inn[i] = yv[i] - ypred[i];             n1 += inn[i] * inn[i];
  }
#pragma unroll
  for (int i = 0; i < MD; ++i) {
    fev[i] = pv[i] - post_prev[b * MD + i];  n2 += fev[i] * fev[i];
    fup[i] = pv[i] - prior_prev[b * MD + i]; n3 += fup[i] * fup[i];
  }
  float s0 = 1.f / (sqrtf(n0) + EPS_C);
  float s1 = 1.f / (sqrtf(n1) + EPS_C);
  float s2 = 1.f / (sqrtf(n2) + EPS_C);
  float s3 = 1.f / (sqrtf(n3) + EPS_C);
  bf16* fr = feats + (size_t)b * KPAD_IN;
#pragma unroll
  for (int i = 0; i < ND; ++i) fr[i] = (bf16)(obs[i] * s0);
#pragma unroll
  for (int i = 0; i < ND; ++i) fr[10 + i] = (bf16)(inn[i] * s1);
#pragma unroll
  for (int i = 0; i < MD; ++i) fr[20 + i] = (bf16)(fev[i] * s2);
#pragma unroll
  for (int i = 0; i < MD; ++i) fr[30 + i] = (bf16)(fup[i] * s3);
#pragma unroll
  for (int i = 40; i < KPAD_IN; ++i) fr[i] = (bf16)0.f;
#pragma unroll
  for (int i = 0; i < MD; ++i) prior_out[b * MD + i] = prior[i];
#pragma unroll
  for (int i = 0; i < ND; ++i) dy_out[b * ND + i] = yv[i] - ypred[i];
}

// ---------------- f32 -> bf16 with zero padding ----------------
__global__ void cvt_pad(const float* __restrict__ src, bf16* __restrict__ dst,
                        int vrows, int vcols, int prows, int pcols) {
  int total = prows * pcols;
  for (int idx = blockIdx.x * blockDim.x + threadIdx.x; idx < total;
       idx += gridDim.x * blockDim.x) {
    int r = idx / pcols, c = idx - r * pcols;
    float v = (r < vrows && c < vcols) ? src[(size_t)r * vcols + c] : 0.0f;
    dst[idx] = (bf16)v;
  }
}

// zero pad columns [HD, KPAD) of a BATCH x KPAD bf16 buffer
__global__ void zero_pad_cols(bf16* __restrict__ buf) {
  int idx = blockIdx.x * blockDim.x + threadIdx.x;
  int padw = KPAD - HD;                       // 16
  int total = BATCH * padw;
  if (idx >= total) return;
  int r = idx / padw, c = HD + (idx - r * padw);
  buf[(size_t)r * KPAD + c] = (bf16)0.f;
}

// ---------------- GEMM1: x = relu(feats @ W_in^T + b_in) -> bf16 padded ----------------
__global__ void gemm_in(const bf16* __restrict__ A,     // 4096 x 64
                        const bf16* __restrict__ W,     // 2000 x 64 (N x K)
                        const float* __restrict__ bias, // 2000
                        bf16* __restrict__ X) {         // 4096 x KPAD
  int lane = threadIdx.x & 31;
  int wave = threadIdx.x >> 5;
  int j0 = blockIdx.x * 16;
  int m0 = blockIdx.y * 128 + wave * 16;
  v8f acc = {};
#pragma unroll
  for (int k = 0; k < KPAD_IN; k += 32) {
    v16bf a = load_fragA(A, KPAD_IN, m0, k, lane);
    v16bf b = load_fragB(W, KPAD_IN, j0, k, lane);
    acc = wmma_bf16(a, b, acc);
  }
  int n = j0 + (lane & 15);
  float bv = bias[n];
  int mo = (lane >> 4) * 8;
#pragma unroll
  for (int i = 0; i < 8; ++i) {
    float v = acc[i] + bv;
    v = v > 0.f ? v : 0.f;
    X[(size_t)(m0 + mo + i) * KPAD + n] = (bf16)v;
  }
}

// ---------------- fused GRU ----------------
// Block = 256 threads (8 waves). Tile: M=256 (32 rows/wave), N=16.
// Weight tiles (6 x 16rows x 32K bf16 = 6 KB) staged to LDS, double-buffered:
// TDM tensor_load_to_lds (3D tile: 32 x 16 x 3 gates) when available.
__global__ void gru_kernel(const bf16* __restrict__ X,    // 4096 x KPAD
                           const bf16* __restrict__ H0,   // 4096 x KPAD
                           const bf16* __restrict__ Wih,  // 6000 x KPAD
                           const bf16* __restrict__ Whh,  // 6000 x KPAD
                           const float* __restrict__ bih, // 6000
                           const float* __restrict__ bhh, // 6000
                           const float* __restrict__ h0f, // 4096 x 2000 (f32)
                           bf16* __restrict__ H1) {       // 4096 x KPAD
  __shared__ __align__(16) bf16 wbuf[2][6][16][32];       // 12 KB
  int lane = threadIdx.x & 31;
  int wave = threadIdx.x >> 5;
  int j0 = blockIdx.x * 16;                 // hidden-unit tile
  int m0 = blockIdx.y * 256 + wave * 32;    // 2 M-subtiles per wave
  v8f acc[6][2];
#pragma unroll
  for (int g = 0; g < 6; ++g)
#pragma unroll
    for (int s = 0; s < 2; ++s) acc[g][s] = (v8f){};

#if HAS_TDM
  unsigned lds_base = (unsigned)(uintptr_t)&wbuf[0][0][0][0];
  if (threadIdx.x < 32) {
    tdm_load_w_tile(Wih + (size_t)j0 * KPAD, lds_base);
    tdm_load_w_tile(Whh + (size_t)j0 * KPAD, lds_base + 3 * 1024u);
    __builtin_amdgcn_s_wait_tensorcnt(0);
  }
  __syncthreads();
#endif

  for (int kk = 0; kk < NKSTEP; ++kk) {
    int k = kk * 32;
#if HAS_TDM
    int cur = kk & 1;
    if (kk + 1 < NKSTEP && threadIdx.x < 32) {
      unsigned nb = lds_base + (unsigned)(cur ^ 1) * (6u * 1024u);
      tdm_load_w_tile(Wih + (size_t)j0 * KPAD + (kk + 1) * 32, nb);
      tdm_load_w_tile(Whh + (size_t)j0 * KPAD + (kk + 1) * 32, nb + 3 * 1024u);
    }
#else
    int cur = 0;
    __syncthreads();
    {
      int t = threadIdx.x;
      if (t < 192) {                       // 192 threads x 32 B = 6 KB
        int f = t >> 5;                    // 0..5
        int r = (t >> 1) & 15;             // row
        int half = t & 1;                  // K half (16 elems)
        int g = (f < 3) ? f : f - 3;
        const bf16* srcm = (f < 3) ? Wih : Whh;
        const v8bf* s = (const v8bf*)(srcm + (size_t)(j0 + g * HD + r) * KPAD +
                                      k + half * 16);
        v8bf* d = (v8bf*)&wbuf[0][f][r][half * 16];
        d[0] = s[0];
        d[1] = s[1];
      }
    }
    __syncthreads();
#endif
    // A fragments: x and h0, two M-subtiles each (global, L2-resident)
    v16bf ax0 = load_fragA(X,  KPAD, m0,      k, lane);
    v16bf ax1 = load_fragA(X,  KPAD, m0 + 16, k, lane);
    v16bf ah0 = load_fragA(H0, KPAD, m0,      k, lane);
    v16bf ah1 = load_fragA(H0, KPAD, m0 + 16, k, lane);
    // B fragments from LDS
#pragma unroll
    for (int g = 0; g < 3; ++g) {
      v16bf wi = lds_fragB(&wbuf[cur][g][0][0], lane);
      acc[g][0] = wmma_bf16(ax0, wi, acc[g][0]);
      acc[g][1] = wmma_bf16(ax1, wi, acc[g][1]);
      v16bf wh = lds_fragB(&wbuf[cur][3 + g][0][0], lane);
      acc[3 + g][0] = wmma_bf16(ah0, wh, acc[3 + g][0]);
      acc[3 + g][1] = wmma_bf16(ah1, wh, acc[3 + g][1]);
    }
#if HAS_TDM
    if (threadIdx.x < 32) __builtin_amdgcn_s_wait_tensorcnt(0);
    __syncthreads();
#endif
  }

  int n = j0 + (lane & 15);
  float bir = bih[n], biz = bih[n + HD], bin_ = bih[n + 2 * HD];
  float bhr = bhh[n], bhz = bhh[n + HD], bhn = bhh[n + 2 * HD];
  int mo = (lane >> 4) * 8;
#pragma unroll
  for (int s = 0; s < 2; ++s) {
#pragma unroll
    for (int i = 0; i < 8; ++i) {
      int m = m0 + s * 16 + mo + i;
      float h0v = h0f[(size_t)m * HD + n];
      float r  = sigmoidf_((acc[0][s][i] + bir) + (acc[3][s][i] + bhr));
      float z  = sigmoidf_((acc[1][s][i] + biz) + (acc[4][s][i] + bhz));
      float nc = tanhf((acc[2][s][i] + bin_) + r * (acc[5][s][i] + bhn));
      float h1 = (1.0f - z) * nc + z * h0v;
      H1[(size_t)m * KPAD + n] = (bf16)h1;
    }
  }
}

// ---------------- GEMM3: kg = tanh(h1 @ W_out^T + b_out) ----------------
__global__ void gemm_out(const bf16* __restrict__ H1,    // 4096 x KPAD
                         const bf16* __restrict__ Wo,    // 112 x KPAD
                         const float* __restrict__ bias, // 100
                         float* __restrict__ KG) {       // 4096 x 100
  int lane = threadIdx.x & 31;
  int wave = threadIdx.x >> 5;
  int j0 = blockIdx.x * 16;                 // up to 112
  int m0 = blockIdx.y * 128 + wave * 16;
  v8f acc = {};
  for (int k = 0; k < KPAD; k += 32) {
    v16bf a = load_fragA(H1, KPAD, m0, k, lane);
    v16bf b = load_fragB(Wo, KPAD, j0, k, lane);
    acc = wmma_bf16(a, b, acc);
  }
  int n = j0 + (lane & 15);
  if (n < MD * ND) {
    float bv = bias[n];
    int mo = (lane >> 4) * 8;
#pragma unroll
    for (int i = 0; i < 8; ++i)
      KG[(size_t)(m0 + mo + i) * (MD * ND) + n] = tanhf(acc[i] + bv);
  }
}

// ---------------- finalize: out = prior + KGain @ dy ----------------
__global__ void finalize_kernel(const float* __restrict__ prior,
                                const float* __restrict__ dy,
                                const float* __restrict__ KG,
                                float* __restrict__ out) {
  int b = blockIdx.x * blockDim.x + threadIdx.x;
  if (b >= BATCH) return;
  float d[ND];
#pragma unroll
  for (int j = 0; j < ND; ++j) d[j] = dy[b * ND + j];
#pragma unroll
  for (int i = 0; i < MD; ++i) {
    float s = prior[b * MD + i];
#pragma unroll
    for (int j = 0; j < ND; ++j) s += KG[b * MD * ND + i * ND + j] * d[j];
    out[b * MD + i] = s;
  }
}

extern "C" void kernel_launch(void* const* d_in, const int* in_sizes, int n_in,
                              void* d_out, int out_size, void* d_ws, size_t ws_size,
                              hipStream_t stream) {
  const float* y        = (const float*)d_in[0];
  const float* post     = (const float*)d_in[1];
  const float* post_prev= (const float*)d_in[2];
  const float* priorprev= (const float*)d_in[3];
  const float* y_prev   = (const float*)d_in[4];
  const float* h_gru    = (const float*)d_in[5];   // (1,B,H)
  const float* F        = (const float*)d_in[6];
  const float* Hm       = (const float*)d_in[7];
  const float* W_in     = (const float*)d_in[8];
  const float* b_in     = (const float*)d_in[9];
  const float* W_ih     = (const float*)d_in[10];
  const float* W_hh     = (const float*)d_in[11];
  const float* b_ih     = (const float*)d_in[12];
  const float* b_hh     = (const float*)d_in[13];
  const float* W_out    = (const float*)d_in[14];
  const float* b_out    = (const float*)d_in[15];
  float* out = (float*)d_out;

  char* ws = (char*)d_ws;
  bf16* featsbf = (bf16*)(ws + OFF_FEATS);
  bf16* winbf   = (bf16*)(ws + OFF_WIN);
  bf16* xbf     = (bf16*)(ws + OFF_X);
  bf16* h0bf    = (bf16*)(ws + OFF_H0);
  bf16* h1bf    = (bf16*)(ws + OFF_H1);
  bf16* wihbf   = (bf16*)(ws + OFF_WIH);
  bf16* whhbf   = (bf16*)(ws + OFF_WHH);
  bf16* woutbf  = (bf16*)(ws + OFF_WOUT);
  float* prior  = (float*)(ws + OFF_PRIOR);
  float* dyb    = (float*)(ws + OFF_DY);
  float* kg     = (float*)(ws + OFF_KG);

  // 1) prep: priors, feats, dy
  prep_kernel<<<BATCH / 256, 256, 0, stream>>>(y, post, post_prev, priorprev,
                                               y_prev, F, Hm, prior, dyb, featsbf);
  // 2) convert / pad operands to bf16
  cvt_pad<<<2048, 256, 0, stream>>>(W_in, winbf, HD, DINF, HD, KPAD_IN);
  cvt_pad<<<4096, 256, 0, stream>>>(h_gru, h0bf, BATCH, HD, BATCH, KPAD);
  cvt_pad<<<4096, 256, 0, stream>>>(W_ih, wihbf, 3 * HD, HD, 3 * HD, KPAD);
  cvt_pad<<<4096, 256, 0, stream>>>(W_hh, whhbf, 3 * HD, HD, 3 * HD, KPAD);
  cvt_pad<<<1024, 256, 0, stream>>>(W_out, woutbf, MD * ND, HD, 112, KPAD);
  zero_pad_cols<<<(BATCH * (KPAD - HD) + 255) / 256, 256, 0, stream>>>(xbf);
  zero_pad_cols<<<(BATCH * (KPAD - HD) + 255) / 256, 256, 0, stream>>>(h1bf);

  // 3) x = relu(feats @ W_in^T + b_in)        (M=4096, N=2000, K=64)
  gemm_in<<<dim3(HD / 16, BATCH / 128), 256, 0, stream>>>(featsbf, winbf, b_in, xbf);

  // 4) fused GRU cell (M=4096, N=2000, K=2016, 6 GEMMs; TDM-staged weights)
  gru_kernel<<<dim3(HD / 16, BATCH / 256), 256, 0, stream>>>(
      xbf, h0bf, wihbf, whhbf, b_ih, b_hh, h_gru, h1bf);

  // 5) kg = tanh(h1 @ W_out^T + b_out)         (M=4096, N=112, K=2016)
  gemm_out<<<dim3(112 / 16, BATCH / 128), 256, 0, stream>>>(h1bf, woutbf, b_out, kg);

  // 6) out = prior + KGain @ dy
  finalize_kernel<<<BATCH / 256, 256, 0, stream>>>(prior, dyb, kg, out);

  (void)in_sizes; (void)n_in; (void)out_size; (void)ws_size;
}
